// HybridMambaBlockLite_23072564314112
// MI455X (gfx1250) — matrix operand order repaired and checked by reading, not compiled, not run on hardware
//
#include <hip/hip_runtime.h>
#include <hip/hip_bf16.h>

typedef _Float16 h16;
typedef __attribute__((ext_vector_type(16))) _Float16 v16h;
typedef __attribute__((ext_vector_type(8)))  _Float16 h8;
typedef __attribute__((ext_vector_type(8)))  float    v8f;

namespace cfg {
constexpr int B = 2, C = 128, H = 128, W = 128;
constexpr int HW = H * W;             // 16384
constexpr int TOKL = B * H * W;       // 32768
constexpr int DIL = 256, DRL = 8;
constexpr int CG = 64;
constexpr int TOKG = B * 64 * 64;     // 8192
constexpr int DIG = 128, DRG = 4;
}

// ---------------- workspace layout (bytes; all 256-aligned) -----------------
constexpr size_t O_W16_FUSE   = 0;          // 128x256 f16
constexpr size_t O_W16_LM_IN  = 65536;      // 512x128
constexpr size_t O_W16_LM_XP  = 196608;     // 48x256 (pad 40->48)
constexpr size_t O_W16_LM_OUT = 221184;     // 128x256
constexpr size_t O_W16_OUT    = 286720;     // 128x128
constexpr size_t O_W16_GIN    = 319488;     // 64x128
constexpr size_t O_W16_GOUT   = 335872;     // 128x64
constexpr size_t O_W16_GM_IN  = 352256;     // 256x64
constexpr size_t O_W16_GM_XP  = 385024;     // 48x128 (pad 36->48)
constexpr size_t O_W16_GM_OUT = 397312;     // 64x128
constexpr size_t O_CAT16  = 413696;                   // 32768x256 f16
constexpr size_t O_XNHWC  = O_CAT16  + 16777216;      // 32768x128 f32
constexpr size_t O_SEQLN  = O_XNHWC  + 16777216;      // 32768x128 f16
constexpr size_t O_XZ     = O_SEQLN  + 8388608;       // 32768x512 f16 (xc|z)
constexpr size_t O_U16    = O_XZ     + 33554432;      // 32768x256 f16 (per-dir)
constexpr size_t O_XDBL   = O_U16    + 16777216;      // 32768x48 f16
constexpr size_t O_DT16   = O_XDBL   + 3145728;       // 32768x256 f16
constexpr size_t O_GACC   = O_DT16   + 16777216;      // 32768x256 f16
constexpr size_t O_DELTA  = O_GACC   + 16777216;      // 32768x128 f32
constexpr size_t O_XG     = O_DELTA  + 16777216;      // 8192x64 f32
constexpr size_t O_GLN    = O_XG     + 2097152;       // 8192x64 f16 (snake order)
constexpr size_t O_XZG    = O_GLN    + 1048576;       // 8192x256 f16
constexpr size_t O_UG     = O_XZG    + 4194304;       // 8192x128 f16
constexpr size_t O_XDBLG  = O_UG     + 2097152;       // 8192x48 f16
constexpr size_t O_DTG    = O_XDBLG  + 786432;        // 8192x128 f16
constexpr size_t O_GACCG  = O_DTG    + 2097152;       // 8192x128 f16
constexpr size_t O_GYSEQ  = O_GACCG  + 2097152;       // 8192x64 f16
constexpr size_t O_GYC    = O_GYSEQ  + 1048576;       // 8192x128 f32
// total ~166 MB of d_ws assumed available.

// ---------------- WMMA fragment helpers (ISA 7.12.2 layouts) ----------------
__device__ __forceinline__ v16h frag_a(const h16* A, int lda, int m0, int k0) {
  int l = threadIdx.x & 31;
  int m = l & 15, hf = l >> 4;
  const h16* p = A + (size_t)(m0 + m) * lda + k0 + hf * 8;
  h8 lo = *(const h8*)p;         // k = half*8 + e      (e=0..7)
  h8 hi = *(const h8*)(p + 16);  // k = 16 + half*8 + e (e=8..15)
  v16h r;
#pragma unroll
  for (int i = 0; i < 8; ++i) { r[i] = lo[i]; r[i + 8] = hi[i]; }
  return r;
}
__device__ __forceinline__ v16h frag_b(const h16* Wt, int K, int n0, int k0) {
  int l = threadIdx.x & 31;
  int n = l & 15, hf = l >> 4;
  const h16* p = Wt + (size_t)(n0 + n) * K + k0 + hf * 8;  // k = half*8 + e, 16 + half*8 + e
  h8 lo = *(const h8*)p;
  h8 hi = *(const h8*)(p + 16);
  v16h r;
#pragma unroll
  for (int i = 0; i < 8; ++i) { r[i] = lo[i]; r[i + 8] = hi[i]; }
  return r;
}
__device__ __forceinline__ v8f wmma32(v16h a, v16h b, v8f c) {
  return __builtin_amdgcn_wmma_f32_16x16x32_f16(false, a, false, b, (short)0, c,
                                                false, false);
}
__device__ __forceinline__ float siluf(float x) { return x / (1.f + __expf(-x)); }
__device__ __forceinline__ int imin(int a, int b) { return a < b ? a : b; }

// D-tile store: lane n=l&15, row m=(l>>4)*8+v
#define STORE_ACC_F32(ptr, ldc, mbase0)                                        \
  {                                                                            \
    int l = threadIdx.x & 31; int n_ = l & 15, mb_ = (l >> 4) * 8;             \
    _Pragma("unroll") for (int v_ = 0; v_ < 8; ++v_)                           \
      (ptr)[(size_t)((mbase0) + mb_ + v_) * (ldc) + n0 + n_] = acc[v_];        \
  }
#define STORE_ACC_F16(ptr, ldc, mbase0)                                        \
  {                                                                            \
    int l = threadIdx.x & 31; int n_ = l & 15, mb_ = (l >> 4) * 8;             \
    _Pragma("unroll") for (int v_ = 0; v_ < 8; ++v_)                           \
      (ptr)[(size_t)((mbase0) + mb_ + v_) * (ldc) + n0 + n_] = (h16)acc[v_];   \
  }

// ---------------- weight convert (f32 -> f16, with zero padding) ------------
__global__ void k_cvt(const float* __restrict__ src, h16* __restrict__ dst,
                      int srows, int scols, int dcols, int total) {
  int i = blockIdx.x * blockDim.x + threadIdx.x;
  if (i >= total) return;
  int r = i / dcols, c = i - r * dcols;
  dst[i] = (r < srows && c < scols) ? (h16)src[r * scols + c] : (h16)0.f;
}

// concat(rad,img) NCHW -> token-major f16 [TOKL][256]
__global__ void k_cat16(const float* __restrict__ rad, const float* __restrict__ img,
                        h16* __restrict__ cat) {
  size_t i = (size_t)blockIdx.x * blockDim.x + threadIdx.x;
  if (i >= (size_t)cfg::B * 256 * cfg::HW) return;
  size_t hw = i & (cfg::HW - 1);
  size_t c2 = (i >> 14) & 255;
  size_t b  = i >> 22;
  const float* s = (c2 < 128) ? rad : img;
  float v = s[(b * 128 + (c2 & 127)) * cfg::HW + hw];
  cat[(b * cfg::HW + hw) * 256 + c2] = (h16)v;
}

// ---------------- generic GEMM: A f16 [M][lda] global, W f16 [N][K] ---------
// block=128 (4 waves); grid.x = M/64, grid.y = N/16
__global__ void k_gemm(const h16* __restrict__ A, int lda,
                       const h16* __restrict__ Wt, int K,
                       float* __restrict__ outf, h16* __restrict__ outh, int ldc) {
  int wave = threadIdx.x >> 5;
  int m0 = (blockIdx.x * 4 + wave) * 16;
  int n0 = blockIdx.y * 16;
  v8f acc = {};
  for (int k0 = 0; k0 < K; k0 += 32)
    acc = wmma32(frag_a(A, lda, m0, k0), frag_b(Wt, K, n0, k0), acc);
  if (outf) { STORE_ACC_F32(outf, ldc, m0); }
  else      { STORE_ACC_F16(outh, ldc, m0); }
}

// ---------------- layernorm: one wave per token -----------------------------
__global__ void k_ln(const float* __restrict__ src, const float* __restrict__ gam,
                     const float* __restrict__ bet, h16* __restrict__ dst,
                     int Cn, int ntok, int snake) {
  int lane = threadIdx.x & 31;
  int wv = blockIdx.x * (blockDim.x >> 5) + (threadIdx.x >> 5);
  if (wv >= ntok) return;
  int tsrc = wv;
  if (snake) {  // dst index wv is sequence order j; source is grid site idx[j]
    int b = wv >> 12, p = wv & 4095;
    int r = p >> 6, c = p & 63;
    tsrc = (b << 12) | (r << 6) | ((r & 1) ? (63 - c) : c);
  }
  const float* row = src + (size_t)tsrc * Cn;
  int per = Cn >> 5;
  float vals[4], s = 0.f, s2 = 0.f;
#pragma unroll
  for (int i = 0; i < 4; ++i)
    if (i < per) { float v = row[lane + 32 * i]; vals[i] = v; s += v; s2 += v * v; }
  for (int off = 16; off; off >>= 1) { s += __shfl_xor(s, off); s2 += __shfl_xor(s2, off); }
  float inv = 1.f / Cn;
  float mu = s * inv;
  float rs = rsqrtf(s2 * inv - mu * mu + 1e-5f);
  h16* drow = dst + (size_t)wv * Cn;
#pragma unroll
  for (int i = 0; i < 4; ++i)
    if (i < per) { int c = lane + 32 * i; drow[c] = (h16)((vals[i] - mu) * rs * gam[c] + bet[c]); }
}

// ---------------- depthwise causal conv (K=4) + SiLU, per direction ---------
__global__ void k_conv(const h16* __restrict__ xz, int ldx, int di, int L,
                       const float* __restrict__ cw, const float* __restrict__ cb,
                       h16* __restrict__ u, int ntok, int dir) {
  int i = blockIdx.x * blockDim.x + threadIdx.x;
  if (i >= ntok * di) return;
  int tok = i / di, d = i - tok * di;
  int p = tok % L;
  float acc = cb[d];
#pragma unroll
  for (int k = 0; k < 4; ++k) {
    int pp = dir ? (p + 3 - k) : (p - 3 + k);
    if (pp >= 0 && pp < L)
      acc += cw[d * 4 + k] * (float)xz[(size_t)(tok + (pp - p)) * ldx + d];
  }
  u[(size_t)tok * di + d] = (h16)(acc * (1.f / (1.f + __expf(-acc))));
}

// ---------------- dt = softplus(x_dbl[:, :dr] @ dt_w.T + dt_b) --------------
__global__ void k_dt(const h16* __restrict__ xdbl, int dr,
                     const float* __restrict__ dtw, const float* __restrict__ dtb,
                     h16* __restrict__ dt, int di, int ntok) {
  int i = blockIdx.x * blockDim.x + threadIdx.x;
  if (i >= ntok * di) return;
  int tok = i / di, d = i - tok * di;
  float a = dtb[d];
  for (int r = 0; r < dr; ++r)
    a += (float)xdbl[(size_t)tok * 48 + r] * dtw[d * dr + r];
  float sp = (a > 20.f) ? a : __logf(1.f + __expf(a));
  dt[(size_t)tok * di + d] = (h16)sp;
}

// ---------------- selective scan: block = one sequence, thread = channel ----
__global__ void k_scan(const h16* __restrict__ u, const h16* __restrict__ dt,
                       const h16* __restrict__ xdbl, int dr,
                       const float* __restrict__ Alog, const float* __restrict__ Dv,
                       h16* __restrict__ gacc, int di, int L, int dir) {
  __shared__ __align__(16) float bc[128 * 32];   // B[16] | C[16] per step
  int d = threadIdx.x;                            // blockDim.x == di
  size_t t0 = (size_t)blockIdx.x * L;
  float As[16];
#pragma unroll
  for (int s = 0; s < 16; ++s) As[s] = -__expf(Alog[d * 16 + s]);
  float Dd = Dv[d];
  float h[16];
#pragma unroll
  for (int s = 0; s < 16; ++s) h[s] = 0.f;

  int nchunk = L / 128;
  for (int cc = 0; cc < nchunk; ++cc) {
    int c0 = dir ? (L - 128 - cc * 128) : (cc * 128);
    __syncthreads();
    for (int e = d; e < 128 * 32; e += blockDim.x) {
      int ii = e >> 5, j = e & 31;
      bc[e] = (float)xdbl[(t0 + c0 + ii) * 48 + dr + j];
    }
    __syncthreads();
    if (cc + 1 < nchunk) {  // hint next chunk of dt/u into cache
      int cn = dir ? (c0 - 128) : (c0 + 128);
      __builtin_prefetch(dt + (t0 + cn) * di + d, 0, 1);
      __builtin_prefetch(u  + (t0 + cn) * di + d, 0, 1);
    }
    for (int st = 0; st < 128; ++st) {
      int ii = dir ? (127 - st) : st;
      size_t tok = t0 + c0 + ii;
      float dtv = (float)dt[tok * di + d];
      float uv  = (float)u[tok * di + d];
      float du = dtv * uv;
      const float* row = &bc[ii * 32];
      float y = 0.f;
#pragma unroll
      for (int s = 0; s < 16; ++s) {
        float w = __expf(dtv * As[s]);
        h[s] = h[s] * w + du * row[s];
        y = fmaf(h[s], row[16 + s], y);
      }
      y += uv * Dd;
      size_t go = tok * di + d;
      if (dir) gacc[go] = (h16)((float)gacc[go] + 0.5f * y);
      else     gacc[go] = (h16)(0.5f * y);
    }
  }
}

// ---------------- gated out-projection: A = gacc * silu(z), staged in LDS ---
// block=128 (4 waves); grid.x = M/64; loops all N tiles after one staging
__global__ void k_gate_gemm(const h16* __restrict__ g, const h16* __restrict__ z,
                            int ldz, const h16* __restrict__ Wt, int K, int N,
                            float* __restrict__ outf, h16* __restrict__ outh, int ldc) {
  __shared__ __align__(16) h16 sA[64 * 256];
  int t0 = blockIdx.x * 64;
  for (int e = threadIdx.x; e < 64 * K; e += blockDim.x) {
    int m = e / K, k = e - m * K;
    size_t tok = t0 + m;
    sA[m * K + k] = (h16)((float)g[tok * K + k] * siluf((float)z[tok * ldz + k]));
  }
  __syncthreads();
  int m0 = (threadIdx.x >> 5) * 16;
  for (int n0 = 0; n0 < N; n0 += 16) {
    v8f acc = {};
    for (int k0 = 0; k0 < K; k0 += 32)
      acc = wmma32(frag_a(sA, K, m0, k0), frag_b(Wt, K, n0, k0), acc);
    if (outf) { STORE_ACC_F32(outf, ldc, t0 + m0); }
    else      { STORE_ACC_F16(outh, ldc, t0 + m0); }
  }
}

// ---------------- 2x2 mean pool + g_in projection (K=128, N=64) -------------
__global__ void k_avg_proj(const float* __restrict__ x, const h16* __restrict__ Wt,
                           float* __restrict__ xg) {
  __shared__ __align__(16) h16 sA[64 * 128];
  int t0 = blockIdx.x * 64;
  for (int e = threadIdx.x; e < 64 * 128; e += blockDim.x) {
    int m = e >> 7, k = e & 127;
    int site = t0 + m;
    int b = site >> 12, p = site & 4095;
    int hs = p >> 6, ws = p & 63;
    size_t base = (size_t)b * cfg::HW + (2 * hs) * cfg::W + 2 * ws;
    float v = 0.25f * (x[base * 128 + k] + x[(base + 1) * 128 + k] +
                       x[(base + cfg::W) * 128 + k] + x[(base + cfg::W + 1) * 128 + k]);
    sA[m * 128 + k] = (h16)v;
  }
  __syncthreads();
  int m0 = (threadIdx.x >> 5) * 16;
  for (int n0 = 0; n0 < 64; n0 += 16) {
    v8f acc = {};
    for (int k0 = 0; k0 < 128; k0 += 32)
      acc = wmma32(frag_a(sA, 128, m0, k0), frag_b(Wt, 128, n0, k0), acc);
    STORE_ACC_F32(xg, 64, t0 + m0);
  }
}

// ---------------- un-snake + g_out projection (K=64, N=128) -----------------
__global__ void k_snake_proj(const h16* __restrict__ gyseq, const h16* __restrict__ Wt,
                             float* __restrict__ gyc) {
  __shared__ __align__(16) h16 sA[64 * 64];
  int t0 = blockIdx.x * 64;
  for (int e = threadIdx.x; e < 64 * 64; e += blockDim.x) {
    int m = e >> 6, k = e & 63;
    int site = t0 + m;
    int b = site >> 12, p = site & 4095;
    int r = p >> 6, c = p & 63;
    int j = (r << 6) | ((r & 1) ? (63 - c) : c);   // inv == idx (involution)
    sA[m * 64 + k] = gyseq[((size_t)(b << 12) + j) * 64 + k];
  }
  __syncthreads();
  int m0 = (threadIdx.x >> 5) * 16;
  for (int n0 = 0; n0 < 128; n0 += 16) {
    v8f acc = {};
    for (int k0 = 0; k0 < 64; k0 += 32)
      acc = wmma32(frag_a(sA, 64, m0, k0), frag_b(Wt, 64, n0, k0), acc);
    STORE_ACC_F32(gyc, 128, t0 + m0);
  }
}

// ---------------- bilinear 64->128 upsample, add into delta -----------------
__global__ void k_ups_add(const float* __restrict__ gyc, float* __restrict__ delta) {
  int i = blockIdx.x * blockDim.x + threadIdx.x;
  if (i >= cfg::TOKL * 128) return;
  int c = i & 127;
  int tok = i >> 7;
  int b = tok >> 14, hw = tok & 16383;
  int hh = hw >> 7, ww = hw & 127;
  float sy = fmaxf(hh * 0.5f - 0.25f, 0.f);
  int y0 = (int)sy; float fy = sy - y0; int y1 = imin(y0 + 1, 63);
  float sx = fmaxf(ww * 0.5f - 0.25f, 0.f);
  int x0 = (int)sx; float fx = sx - x0; int x1 = imin(x0 + 1, 63);
  size_t base = (size_t)b << 12;
  float v00 = gyc[((base + y0 * 64 + x0) << 7) + c];
  float v01 = gyc[((base + y0 * 64 + x1) << 7) + c];
  float v10 = gyc[((base + y1 * 64 + x0) << 7) + c];
  float v11 = gyc[((base + y1 * 64 + x1) << 7) + c];
  delta[i] += (v00 * (1.f - fx) + v01 * fx) * (1.f - fy) + (v10 * (1.f - fx) + v11 * fx) * fy;
}

// ---------------- final projection, NCHW store ------------------------------
__global__ void k_final(const float* __restrict__ delta, const h16* __restrict__ Wt,
                        float* __restrict__ out) {
  __shared__ __align__(16) h16 sA[64 * 128];
  int t0 = blockIdx.x * 64;
  for (int e = threadIdx.x; e < 64 * 128; e += blockDim.x) {
    int m = e >> 7, k = e & 127;
    sA[m * 128 + k] = (h16)delta[(size_t)(t0 + m) * 128 + k];
  }
  __syncthreads();
  int m0 = (threadIdx.x >> 5) * 16;
  for (int n0 = 0; n0 < 128; n0 += 16) {
    v8f acc = {};
    for (int k0 = 0; k0 < 128; k0 += 32)
      acc = wmma32(frag_a(sA, 128, m0, k0), frag_b(Wt, 128, n0, k0), acc);
    int l = threadIdx.x & 31; int n = l & 15, mb = (l >> 4) * 8;
#pragma unroll
    for (int v = 0; v < 8; ++v) {
      int tok = t0 + m0 + mb + v;
      int b = tok >> 14, hw = tok & 16383;
      out[((size_t)(b * 128 + n0 + n) << 14) + hw] = acc[v];
    }
  }
}

// ---------------------------------------------------------------------------
extern "C" void kernel_launch(void* const* d_in, const int* in_sizes, int n_in,
                              void* d_out, int out_size, void* d_ws, size_t ws_size,
                              hipStream_t stream) {
  (void)in_sizes; (void)n_in; (void)out_size; (void)ws_size;
  const float* rad      = (const float*)d_in[0];
  const float* img      = (const float*)d_in[1];
  const float* fuse_w   = (const float*)d_in[2];
  const float* g_in_w   = (const float*)d_in[3];
  const float* g_out_w  = (const float*)d_in[4];
  const float* out_w    = (const float*)d_in[5];
  const float* ln_l_g   = (const float*)d_in[6];
  const float* ln_l_b   = (const float*)d_in[7];
  const float* ln_g_g   = (const float*)d_in[8];
  const float* ln_g_b   = (const float*)d_in[9];
  const float* lm_in_w  = (const float*)d_in[10];
  const float* lm_cw    = (const float*)d_in[11];
  const float* lm_cb    = (const float*)d_in[12];
  const float* lm_xp    = (const float*)d_in[13];
  const float* lm_dtw   = (const float*)d_in[14];
  const float* lm_dtb   = (const float*)d_in[15];
  const float* lm_Alog  = (const float*)d_in[16];
  const float* lm_D     = (const float*)d_in[17];
  const float* lm_out_w = (const float*)d_in[18];
  const float* gm_in_w  = (const float*)d_in[19];
  const float* gm_cw    = (const float*)d_in[20];
  const float* gm_cb    = (const float*)d_in[21];
  const float* gm_xp    = (const float*)d_in[22];
  const float* gm_dtw   = (const float*)d_in[23];
  const float* gm_dtb   = (const float*)d_in[24];
  const float* gm_Alog  = (const float*)d_in[25];
  const float* gm_D     = (const float*)d_in[26];
  const float* gm_out_w = (const float*)d_in[27];

  char* ws = (char*)d_ws;
  h16* w16_fuse   = (h16*)(ws + O_W16_FUSE);
  h16* w16_lm_in  = (h16*)(ws + O_W16_LM_IN);
  h16* w16_lm_xp  = (h16*)(ws + O_W16_LM_XP);
  h16* w16_lm_out = (h16*)(ws + O_W16_LM_OUT);
  h16* w16_out    = (h16*)(ws + O_W16_OUT);
  h16* w16_gin    = (h16*)(ws + O_W16_GIN);
  h16* w16_gout   = (h16*)(ws + O_W16_GOUT);
  h16* w16_gm_in  = (h16*)(ws + O_W16_GM_IN);
  h16* w16_gm_xp  = (h16*)(ws + O_W16_GM_XP);
  h16* w16_gm_out = (h16*)(ws + O_W16_GM_OUT);
  h16*   cat16 = (h16*)(ws + O_CAT16);
  float* xnhwc = (float*)(ws + O_XNHWC);
  h16*   seqln = (h16*)(ws + O_SEQLN);
  h16*   xz    = (h16*)(ws + O_XZ);
  h16*   u16   = (h16*)(ws + O_U16);
  h16*   xdbl  = (h16*)(ws + O_XDBL);
  h16*   dt16  = (h16*)(ws + O_DT16);
  h16*   gacc  = (h16*)(ws + O_GACC);
  float* delta = (float*)(ws + O_DELTA);
  float* xg    = (float*)(ws + O_XG);
  h16*   gln   = (h16*)(ws + O_GLN);
  h16*   xzg   = (h16*)(ws + O_XZG);
  h16*   ug    = (h16*)(ws + O_UG);
  h16*   xdblg = (h16*)(ws + O_XDBLG);
  h16*   dtg   = (h16*)(ws + O_DTG);
  h16*   gaccg = (h16*)(ws + O_GACCG);
  h16*   gyseq = (h16*)(ws + O_GYSEQ);
  float* gyc   = (float*)(ws + O_GYC);

  auto cvt = [&](const float* s, h16* d, int sr, int sc, int dr_, int dc) {
    int tot = dr_ * dc;
    k_cvt<<<(tot + 255) / 256, 256, 0, stream>>>(s, d, sr, sc, dc, tot);
  };
  cvt(fuse_w,   w16_fuse,   128, 256, 128, 256);
  cvt(lm_in_w,  w16_lm_in,  512, 128, 512, 128);
  cvt(lm_xp,    w16_lm_xp,   40, 256,  48, 256);
  cvt(lm_out_w, w16_lm_out, 128, 256, 128, 256);
  cvt(out_w,    w16_out,    128, 128, 128, 128);
  cvt(g_in_w,   w16_gin,     64, 128,  64, 128);
  cvt(g_out_w,  w16_gout,   128,  64, 128,  64);
  cvt(gm_in_w,  w16_gm_in,  256,  64, 256,  64);
  cvt(gm_xp,    w16_gm_xp,   36, 128,  48, 128);
  cvt(gm_out_w, w16_gm_out,  64, 128,  64, 128);

  // fuse
  k_cat16<<<(cfg::B * 256 * cfg::HW + 255) / 256, 256, 0, stream>>>(rad, img, cat16);
  k_gemm<<<dim3(cfg::TOKL / 64, 128 / 16), 128, 0, stream>>>(cat16, 256, w16_fuse, 256,
                                                             xnhwc, nullptr, 128);
  // local branch
  k_ln<<<cfg::TOKL / 8, 256, 0, stream>>>(xnhwc, ln_l_g, ln_l_b, seqln, 128, cfg::TOKL, 0);
  k_gemm<<<dim3(cfg::TOKL / 64, 512 / 16), 128, 0, stream>>>(seqln, 128, w16_lm_in, 128,
                                                             nullptr, xz, 512);
  for (int dir = 0; dir < 2; ++dir) {
    k_conv<<<(cfg::TOKL * cfg::DIL + 255) / 256, 256, 0, stream>>>(
        xz, 512, cfg::DIL, cfg::W, lm_cw, lm_cb, u16, cfg::TOKL, dir);
    k_gemm<<<dim3(cfg::TOKL / 64, 48 / 16), 128, 0, stream>>>(u16, 256, w16_lm_xp, 256,
                                                              nullptr, xdbl, 48);
    k_dt<<<(cfg::TOKL * cfg::DIL + 255) / 256, 256, 0, stream>>>(
        xdbl, cfg::DRL, lm_dtw, lm_dtb, dt16, cfg::DIL, cfg::TOKL);
    k_scan<<<cfg::TOKL / cfg::W, cfg::DIL, 0, stream>>>(
        u16, dt16, xdbl, cfg::DRL, lm_Alog, lm_D, gacc, cfg::DIL, cfg::W, dir);
  }
  k_gate_gemm<<<cfg::TOKL / 64, 128, 0, stream>>>(gacc, xz + 256, 512, w16_lm_out,
                                                  256, 128, delta, nullptr, 128);
  // global branch
  k_avg_proj<<<cfg::TOKG / 64, 128, 0, stream>>>(xnhwc, w16_gin, xg);
  k_ln<<<cfg::TOKG / 8, 256, 0, stream>>>(xg, ln_g_g, ln_g_b, gln, 64, cfg::TOKG, 1);
  k_gemm<<<dim3(cfg::TOKG / 64, 256 / 16), 128, 0, stream>>>(gln, 64, w16_gm_in, 64,
                                                             nullptr, xzg, 256);
  for (int dir = 0; dir < 2; ++dir) {
    k_conv<<<(cfg::TOKG * cfg::DIG + 255) / 256, 256, 0, stream>>>(
        xzg, 256, cfg::DIG, 4096, gm_cw, gm_cb, ug, cfg::TOKG, dir);
    k_gemm<<<dim3(cfg::TOKG / 64, 48 / 16), 128, 0, stream>>>(ug, 128, w16_gm_xp, 128,
                                                              nullptr, xdblg, 48);
    k_dt<<<(cfg::TOKG * cfg::DIG + 255) / 256, 256, 0, stream>>>(
        xdblg, cfg::DRG, gm_dtw, gm_dtb, dtg, cfg::DIG, cfg::TOKG);
    k_scan<<<cfg::B, cfg::DIG, 0, stream>>>(ug, dtg, xdblg, cfg::DRG, gm_Alog, gm_D,
                                            gaccg, cfg::DIG, 4096, dir);
  }
  k_gate_gemm<<<cfg::TOKG / 64, 128, 0, stream>>>(gaccg, xzg + 128, 256, w16_gm_out,
                                                  128, 64, nullptr, gyseq, 64);
  k_snake_proj<<<cfg::TOKG / 64, 128, 0, stream>>>(gyseq, w16_gout, gyc);
  k_ups_add<<<(cfg::TOKL * 128 + 255) / 256, 256, 0, stream>>>(gyc, delta);
  // final projection
  k_final<<<cfg::TOKL / 64, 128, 0, stream>>>(delta, w16_out, (float*)d_out);
}
